// SpectralAttention_43310450213235
// MI455X (gfx1250) — matrix operand order repaired and checked by reading, not compiled, not run on hardware
//
#include <hip/hip_runtime.h>
#include <cstdint>

// ---------------------------------------------------------------------------
// Types
// ---------------------------------------------------------------------------
typedef __attribute__((ext_vector_type(16))) __bf16 v16bf;
typedef __attribute__((ext_vector_type(8)))  float  v8f;
typedef __attribute__((ext_vector_type(4)))  int    v4i_t;
typedef unsigned short u16;
typedef unsigned int   u32;

union FragBF16 { v16bf v; u32 u[8]; };

// Problem constants (from the reference)
constexpr int SA_B = 4;
constexpr int SA_L = 4096;
constexpr int SA_D = 2048;
constexpr int NCOL = SA_B * SA_D;          // 8192 independent scan columns
constexpr int NSEG = 16;                   // segments per column
constexpr int SEG  = SA_L / NSEG;          // 256
constexpr float SA_SCALE = 0.08838834764831845f;   // 128^-0.5

// ---------------------------------------------------------------------------
// CDNA5 async global->LDS path (ASYNCcnt tracked). Gated on __has_builtin so
// we fall back to register staging if absent. Probe round 2 showed the b128
// builtin expects AS1 pointers to int4 (printed as "int __vector__ __device__*").
// ---------------------------------------------------------------------------
#if defined(__has_builtin)
#  if __has_builtin(__builtin_amdgcn_global_load_async_to_lds_b128) && \
      __has_builtin(__builtin_amdgcn_s_wait_asynccnt)
#    define SA_ASYNC 1
#  endif
#endif
#ifndef SA_ASYNC
#  define SA_ASYNC 0
#endif

#if SA_ASYNC
typedef __attribute__((address_space(1))) v4i_t gv4i;   // global int4
typedef __attribute__((address_space(3))) v4i_t lv4i;   // LDS int4
__device__ __forceinline__ void async_cp_b128(const void* g, const void* l) {
  // generic->AS1 is value-identical; generic->AS3 is low-32-bit LDS offset.
  __builtin_amdgcn_global_load_async_to_lds_b128(
      (gv4i*)(unsigned long long)g,
      (lv4i*)(unsigned int)(unsigned long long)l,
      /*imm offset*/0, /*cpol*/0);
}
#endif

// ---------------------------------------------------------------------------
// bf16 helpers (bit-level, no reliance on scalar __bf16 arithmetic)
// ---------------------------------------------------------------------------
__device__ __forceinline__ u16 f2bf(float f) {
  u32 u = __float_as_uint(f);
  u32 r = ((u >> 16) & 1u) + 0x7FFFu;     // round-to-nearest-even
  return (u16)((u + r) >> 16);
}
__device__ __forceinline__ float bf2f(u16 h) {
  return __uint_as_float(((u32)h) << 16);
}

// ---------------------------------------------------------------------------
// fp32 -> bf16 conversion
// ---------------------------------------------------------------------------
__global__ void cvt_f32_to_bf16(const float* __restrict__ in,
                                u16* __restrict__ out, int n) {
  int i = blockIdx.x * blockDim.x + threadIdx.x;
  if (i < n) out[i] = f2bf(in[i]);
}

// ---------------------------------------------------------------------------
// TN bf16 GEMM with f32 accumulation:
//   out[m, n] = sum_k A[m, k] * W[n, k] + bias[n]
// Block tile: 128 (M) x 256 (N), 256 threads = 8 waves (wave32).
// Wave grid 2(M) x 4(N); each wave: 64x64 = 4x4 WMMA 16x16 tiles
// (16 v_wmma per K-step, 1.0 ds_load_b128 per wmma).
// K-loop step 32 with double-buffered LDS staging (async-LDS if available).
// mode: 0 = bf16 store, 1 = bf16 store with phi=elu(.)+1, 2 = fp32 store
// ---------------------------------------------------------------------------
__global__ __launch_bounds__(256)
void gemm_bf16_tn(const u16* __restrict__ A, const u16* __restrict__ W,
                  const float* __restrict__ bias, void* __restrict__ out,
                  int M, int N, int K, int mode) {
  // Padded LDS tiles: rows x 32 halves, row stride 40 halves (= 20 uints,
  // 80B: 16B-aligned b128 accesses, conflict-free 16-row fragment reads).
  __shared__ u32 lAs[2][128 * 20];   // 20 KB
  __shared__ u32 lBs[2][256 * 20];   // 40 KB

  const int tid  = threadIdx.x;
  const int lane = tid & 31;
  const int wid  = tid >> 5;
  const int wm   = wid >> 2;          // 0..1  (M direction, 64 rows each)
  const int wn   = wid & 3;           // 0..3  (N direction, 64 cols each)
  const int hi   = lane >> 4;         // lane half-group (WMMA layout)
  const int lo   = lane & 15;

  const int m0 = blockIdx.y * 128;
  const int n0 = blockIdx.x * 256;

  // Staging ownership:
  //  A tile (128 x 32 halves): thread t -> row t>>1, half-chunk t&1 (32B).
  //  B tile (256 x 32 halves): thread t -> full row t (64B).
  const size_t arow = (size_t)(m0 + (tid >> 1)) * K + (size_t)(tid & 1) * 16;
  const size_t brow = (size_t)(n0 + tid) * K;
  const int aLds = (tid >> 1) * 20 + (tid & 1) * 8;
  const int bLds = tid * 20;
  const int KT = K / 32;

#if SA_ASYNC
  auto issue_tile = [&](int kt, int buf) {
    const u16* ga = A + arow + (size_t)kt * 32;
    u32* da = &lAs[buf][aLds];
    async_cp_b128(ga,     da);
    async_cp_b128(ga + 8, da + 4);
    const u16* gb = W + brow + (size_t)kt * 32;
    u32* db = &lBs[buf][bLds];
    async_cp_b128(gb,      db);
    async_cp_b128(gb + 8,  db + 4);
    async_cp_b128(gb + 16, db + 8);
    async_cp_b128(gb + 24, db + 12);
  };
  issue_tile(0, 0);
  __builtin_amdgcn_s_wait_asynccnt(0);
#else
  uint4 ta0, ta1, tb0, tb1, tb2, tb3;
  {
    const uint4* pa = reinterpret_cast<const uint4*>(A + arow);
    ta0 = pa[0]; ta1 = pa[1];
    const uint4* pb = reinterpret_cast<const uint4*>(W + brow);
    tb0 = pb[0]; tb1 = pb[1]; tb2 = pb[2]; tb3 = pb[3];
    uint4* da = reinterpret_cast<uint4*>(&lAs[0][aLds]);
    da[0] = ta0; da[1] = ta1;
    uint4* db = reinterpret_cast<uint4*>(&lBs[0][bLds]);
    db[0] = tb0; db[1] = tb1; db[2] = tb2; db[3] = tb3;
  }
#endif
  __syncthreads();

  v8f acc[4][4] = {};

  for (int kt = 0; kt < KT; ++kt) {
    const int cur = kt & 1;
    const int nxt = cur ^ 1;

#if SA_ASYNC
    if (kt + 1 < KT) issue_tile(kt + 1, nxt);   // async engine fills nxt
#else
    if (kt + 1 < KT) {                          // loads in flight over compute
      const uint4* pa =
          reinterpret_cast<const uint4*>(A + arow + (size_t)(kt + 1) * 32);
      ta0 = pa[0]; ta1 = pa[1];
      const uint4* pb =
          reinterpret_cast<const uint4*>(W + brow + (size_t)(kt + 1) * 32);
      tb0 = pb[0]; tb1 = pb[1]; tb2 = pb[2]; tb3 = pb[3];
    }
    if (kt + 2 < KT) {
      __builtin_prefetch(A + arow + (size_t)(kt + 2) * 32, 0, 0);
      __builtin_prefetch(W + brow + (size_t)(kt + 2) * 32, 0, 0);
    }
#endif

    // Assemble WMMA fragments from LDS per the CDNA5 16-bit layouts.
    // A 16x32: lane holds row lo; K-halves {kb..kb+7, kb+16..kb+23}, kb=8*hi.
    // B 32x16: lane holds col lo; K-halves {kb..kb+7, kb+16..kb+23}, kb=8*hi.
    FragBF16 af[4], bfm[4];
    const u32* Ab = lAs[cur];
    const u32* Bb = lBs[cur];
#pragma unroll
    for (int ai = 0; ai < 4; ++ai) {
      const int base = (wm * 64 + ai * 16 + lo) * 20 + hi * 4;
#pragma unroll
      for (int j = 0; j < 4; ++j) {
        af[ai].u[j]     = Ab[base + j];
        af[ai].u[4 + j] = Ab[base + 8 + j];
      }
    }
#pragma unroll
    for (int bj = 0; bj < 4; ++bj) {
      const int base = (wn * 64 + bj * 16 + lo) * 20 + hi * 4;
#pragma unroll
      for (int j = 0; j < 8; ++j) bfm[bj].u[j] = Bb[base + (j & 3) + ((j >> 2) << 3)];
    }

#pragma unroll
    for (int ai = 0; ai < 4; ++ai)
#pragma unroll
      for (int bj = 0; bj < 4; ++bj)
        acc[ai][bj] = __builtin_amdgcn_wmma_f32_16x16x32_bf16(
            false, af[ai].v, false, bfm[bj].v, (short)0, acc[ai][bj],
            false, false);

#if SA_ASYNC
    if (kt + 1 < KT) __builtin_amdgcn_s_wait_asynccnt(0);
#else
    if (kt + 1 < KT) {
      uint4* da = reinterpret_cast<uint4*>(&lAs[nxt][aLds]);
      da[0] = ta0; da[1] = ta1;
      uint4* db = reinterpret_cast<uint4*>(&lBs[nxt][bLds]);
      db[0] = tb0; db[1] = tb1; db[2] = tb2; db[3] = tb3;
    }
#endif
    __syncthreads();
  }

  // Epilogue: C/D layout — lane(l), vgpr(r) -> (M = r + 8*(l>>4), N = l&15).
#pragma unroll
  for (int ai = 0; ai < 4; ++ai) {
    const int mrow = m0 + wm * 64 + ai * 16 + hi * 8;
#pragma unroll
    for (int bj = 0; bj < 4; ++bj) {
      const int nn = n0 + wn * 64 + bj * 16 + lo;
      const float bval = bias[nn];
#pragma unroll
      for (int rr = 0; rr < 8; ++rr) {
        float val = acc[ai][bj][rr] + bval;
        if (mode == 1) val = (val > 0.0f) ? (val + 1.0f) : __expf(val);
        const size_t oi = (size_t)(mrow + rr) * N + nn;
        if (mode == 2) reinterpret_cast<float*>(out)[oi] = val;
        else           reinterpret_cast<u16*>(out)[oi]   = f2bf(val);
      }
    }
  }
}

// ---------------------------------------------------------------------------
// Segmented elementwise prefix sum over L (per (b, d) column).
// Pass 1: per-segment totals of phi_k * v.
// ---------------------------------------------------------------------------
__global__ void scan_partials(const u16* __restrict__ pk,
                              const u16* __restrict__ pv,
                              float* __restrict__ segtot) {
  const int gid = blockIdx.x * blockDim.x + threadIdx.x;   // < NCOL*NSEG
  const int c = gid & (NCOL - 1);
  const int s = gid >> 13;                                 // NCOL == 2^13
  const int b = c / SA_D;
  const int d = c % SA_D;
  size_t idx = ((size_t)b * SA_L + (size_t)s * SEG) * SA_D + d;
  float acc = 0.0f;
  for (int i = 0; i < SEG; ++i, idx += SA_D)
    acc += bf2f(pk[idx]) * bf2f(pv[idx]);
  segtot[(size_t)s * NCOL + c] = acc;
}

// Pass 2: add exclusive prefix of segment totals, emit context = phi_q*kv*scale
__global__ void scan_apply(const u16* __restrict__ pk,
                           const u16* __restrict__ pv,
                           const u16* __restrict__ pq,
                           const float* __restrict__ segtot,
                           u16* __restrict__ ctx) {
  const int gid = blockIdx.x * blockDim.x + threadIdx.x;
  const int c = gid & (NCOL - 1);
  const int s = gid >> 13;
  const int b = c / SA_D;
  const int d = c % SA_D;
  float acc = 0.0f;
  for (int s2 = 0; s2 < s; ++s2) acc += segtot[(size_t)s2 * NCOL + c];
  size_t idx = ((size_t)b * SA_L + (size_t)s * SEG) * SA_D + d;
  for (int i = 0; i < SEG; ++i, idx += SA_D) {
    acc += bf2f(pk[idx]) * bf2f(pv[idx]);
    ctx[idx] = f2bf(bf2f(pq[idx]) * acc * SA_SCALE);
  }
}

// ---------------------------------------------------------------------------
// Host side
// ---------------------------------------------------------------------------
extern "C" void kernel_launch(void* const* d_in, const int* in_sizes, int n_in,
                              void* d_out, int out_size, void* d_ws,
                              size_t ws_size, hipStream_t stream) {
  (void)in_sizes; (void)n_in; (void)out_size; (void)ws_size;

  const float* x  = (const float*)d_in[0];
  const float* Wq = (const float*)d_in[1];
  const float* bq = (const float*)d_in[2];
  const float* Wk = (const float*)d_in[3];
  const float* bk = (const float*)d_in[4];
  const float* Wv = (const float*)d_in[5];
  const float* bv = (const float*)d_in[6];
  const float* Wo = (const float*)d_in[7];
  const float* bo = (const float*)d_in[8];

  const int M = SA_B * SA_L;   // 16384
  const int D = SA_D;          // 2048

  // Workspace carve-up (~353 MB total)
  char* ws = (char*)d_ws;
  size_t off = 0;
  auto carve = [&](size_t bytes) {
    char* p = ws + off;
    off += (bytes + 255) & ~(size_t)255;
    return p;
  };
  u16* xbf  = (u16*)carve((size_t)M * D * 2);
  u16* wqb  = (u16*)carve((size_t)D * D * 2);
  u16* wkb  = (u16*)carve((size_t)D * D * 2);
  u16* wvb  = (u16*)carve((size_t)D * D * 2);
  u16* wob  = (u16*)carve((size_t)D * D * 2);
  u16* phiq = (u16*)carve((size_t)M * D * 2);
  u16* phik = (u16*)carve((size_t)M * D * 2);
  u16* vbf  = (u16*)carve((size_t)M * D * 2);
  u16* ctx  = (u16*)carve((size_t)M * D * 2);
  float* segtot = (float*)carve((size_t)NSEG * NCOL * 4);

  const int nx = M * D;   // 33,554,432
  const int nw = D * D;   //  4,194,304
  cvt_f32_to_bf16<<<(nx + 255) / 256, 256, 0, stream>>>(x,  xbf, nx);
  cvt_f32_to_bf16<<<(nw + 255) / 256, 256, 0, stream>>>(Wq, wqb, nw);
  cvt_f32_to_bf16<<<(nw + 255) / 256, 256, 0, stream>>>(Wk, wkb, nw);
  cvt_f32_to_bf16<<<(nw + 255) / 256, 256, 0, stream>>>(Wv, wvb, nw);
  cvt_f32_to_bf16<<<(nw + 255) / 256, 256, 0, stream>>>(Wo, wob, nw);

  dim3 g(D / 256, M / 128);  // (8, 128)
  gemm_bf16_tn<<<g, 256, 0, stream>>>(xbf, wqb, bq, phiq, M, D, D, 1);
  gemm_bf16_tn<<<g, 256, 0, stream>>>(xbf, wkb, bk, phik, M, D, D, 1);
  gemm_bf16_tn<<<g, 256, 0, stream>>>(xbf, wvb, bv, vbf,  M, D, D, 0);

  const int nscan = NCOL * NSEG;  // 131072
  scan_partials<<<nscan / 256, 256, 0, stream>>>(phik, vbf, segtot);
  scan_apply<<<nscan / 256, 256, 0, stream>>>(phik, vbf, phiq, segtot, ctx);

  gemm_bf16_tn<<<g, 256, 0, stream>>>(ctx, wob, bo, d_out, M, D, D, 2);
}
